// Attention_new_28570122453037
// MI455X (gfx1250) — compile-verified, matrix operand order checked by reading, not run on hardware
//
#include <hip/hip_runtime.h>
#include <hip/hip_bf16.h>
#include <math.h>

#define DIM 256
#define NN  128
#define BB  16
#define ROWS (BB * NN)   // 2048

typedef __attribute__((ext_vector_type(16))) __bf16 v16bf;
typedef __attribute__((ext_vector_type(4)))  __bf16 v4bf;
typedef __attribute__((ext_vector_type(8)))  float  v8f;
typedef __attribute__((ext_vector_type(4)))  float  vf4;
typedef __attribute__((ext_vector_type(4)))  unsigned int vu4;

// Load a 16x32 bf16 WMMA fragment chunk pair for this lane.
// Per CDNA5 ISA (CDNA5 ISA): lane L -> row = L&15, half = L>>4,
// VGPR0..3 hold K = half*8 .. half*8+7, VGPR4..7 hold K = 16+half*8 .. +7.
// p points at element (row, kbase + half*8); second chunk is +16 elements.
__device__ __forceinline__ v16bf load_frag(const __bf16* p) {
    union { v16bf v; struct { vu4 a, b; } s; } u;
    u.s.a = *reinterpret_cast<const vu4*>(p);
    u.s.b = *reinterpret_cast<const vu4*>(p + 16);
    return u.v;
}

// ---------------------------------------------------------------------------
// Kernel 0: weight prep. Cast We/Woe to bf16 (row-major [co][ci] == [N][K],
// which makes B fragments contiguous). Transpose Wq/Wk/Wv/Won to fp32 [ci][co]
// for coalesced GEMV kernels.
// ---------------------------------------------------------------------------
__global__ void __launch_bounds__(DIM) prep_kernel(
    const float* __restrict__ Wq, const float* __restrict__ Wk,
    const float* __restrict__ Wv, const float* __restrict__ Won,
    const float* __restrict__ We, const float* __restrict__ Woe,
    float* __restrict__ WqT, float* __restrict__ WkT,
    float* __restrict__ WvT, float* __restrict__ WonT,
    __bf16* __restrict__ Webf, __bf16* __restrict__ Woebf)
{
    int co = blockIdx.x, ci = threadIdx.x;
    int idx  = co * DIM + ci;
    int tidx = ci * DIM + co;
    Webf[idx]  = (__bf16)We[idx];
    Woebf[idx] = (__bf16)Woe[idx];
    WqT[tidx]  = Wq[idx];
    WkT[tidx]  = Wk[idx];
    WvT[tidx]  = Wv[idx];
    WonT[tidx] = Won[idx];
}

// ---------------------------------------------------------------------------
// Kernel 1: q/k/v projections. One block per node row (2048), thread = out chan.
// ---------------------------------------------------------------------------
__global__ void __launch_bounds__(DIM) qkv_kernel(
    const float* __restrict__ node,
    const float* __restrict__ WqT, const float* __restrict__ bq,
    const float* __restrict__ WkT, const float* __restrict__ bk,
    const float* __restrict__ WvT, const float* __restrict__ bv,
    float* __restrict__ qws, float* __restrict__ kws, float* __restrict__ vws)
{
    __shared__ float sn[DIM];
    int r = blockIdx.x, c = threadIdx.x;
    sn[c] = node[(size_t)r * DIM + c];
    __syncthreads();
    float aq = bq[c], ak = bk[c], av = bv[c];
    #pragma unroll 4
    for (int cc = 0; cc < DIM; ++cc) {
        float x = sn[cc];
        aq += x * WqT[cc * DIM + c];
        ak += x * WkT[cc * DIM + c];
        av += x * WvT[cc * DIM + c];
    }
    qws[(size_t)r * DIM + c] = aq;
    kws[(size_t)r * DIM + c] = ak;
    vws[(size_t)r * DIM + c] = av;
}

// ---------------------------------------------------------------------------
// Kernel 2: fused main kernel. One workgroup per (b,i): 256 threads = 8 waves,
// wave w owns j-tile rows [16w, 16w+16). Both weight matrices stay LDS-resident
// (2x128KB) next to the 64KB edge/attn tile = 320KB, the per-WG LDS maximum.
//   GEMM1: E = edge_row(bf16, LDS) x We^T(bf16, LDS) -> fp32, +be
//   epilogue: attn = q*k/sqrt(d) * (E+1)*E  -> bf16 back into the LDS tile
//   GEMM2 (no barrier needed: own rows only, DS in-order): edge_out -> HBM
//   single barrier, then softmax over j per channel + sum_j a*v -> node_agg
// ---------------------------------------------------------------------------
__global__ void __launch_bounds__(DIM) main_kernel(
    const float*  __restrict__ edge,
    const float*  __restrict__ qws,  const float* __restrict__ kws,
    const float*  __restrict__ vws,
    const float*  __restrict__ be,   const float* __restrict__ boe,
    const __bf16* __restrict__ Webf, const __bf16* __restrict__ Woebf,
    float* __restrict__ aggws, float* __restrict__ edge_out)
{
    extern __shared__ __align__(16) char smem[];
    __bf16* smW1 = (__bf16*)smem;                      // We  bf16: 128KB
    __bf16* smW2 = (__bf16*)(smem + 131072);           // Woe bf16: 128KB
    __bf16* smE  = (__bf16*)(smem + 262144);           // 128x256 bf16: 64KB

    const int bi   = blockIdx.x;        // b*N + i
    const int b    = bi >> 7;           // N = 128
    const int tid  = threadIdx.x;
    const int wv   = tid >> 5;          // wave id (wave32)
    const int lane = tid & 31;
    const int half = lane >> 4;
    const int nlo  = lane & 15;
    const int m0   = wv * 16;

    // ---- stage both weight matrices (bf16) into LDS
    {
        const vu4* s1 = (const vu4*)Webf;
        const vu4* s2 = (const vu4*)Woebf;
        vu4* d1 = (vu4*)smW1;
        vu4* d2 = (vu4*)smW2;
        for (int it = tid; it < 8192; it += DIM) { d1[it] = s1[it]; d2[it] = s2[it]; }
    }
    // ---- stage edge row [128][256] fp32 -> bf16 LDS
    {
        const vf4* src = (const vf4*)(edge + (size_t)bi * (NN * DIM));
        v4bf* dst = (v4bf*)smE;
        for (int it = tid; it < 8192; it += DIM) {
            vf4 f = src[it];
            v4bf h;
            h[0] = (__bf16)f[0]; h[1] = (__bf16)f[1];
            h[2] = (__bf16)f[2]; h[3] = (__bf16)f[3];
            dst[it] = h;
        }
    }
    __syncthreads();

    const float inv_sqrt_dk = 0.17677669529663689f;    // 1/sqrt(32)
    const float* krow = kws + (size_t)b * NN * DIM;
    const float* qrow = qws + (size_t)bi * DIM;

    // ---- GEMM1: cache A fragments (edge rows) for all 8 K-steps
    v16bf afr[8];
    #pragma unroll
    for (int kk = 0; kk < 8; ++kk)
        afr[kk] = load_frag(smE + (m0 + nlo) * DIM + kk * 32 + half * 8);

    #pragma unroll 2
    for (int nt = 0; nt < 16; ++nt) {
        // prefetch all 8 B fragments first: 16 ds_load_b128 in flight,
        // then an uninterrupted 8-deep WMMA accumulation burst
        v16bf bfr[8];
        #pragma unroll
        for (int kk = 0; kk < 8; ++kk)
            bfr[kk] = load_frag(smW1 + (nt * 16 + nlo) * DIM + kk * 32 + half * 8);
        v8f acc = {};
        #pragma unroll
        for (int kk = 0; kk < 8; ++kk)
            acc = __builtin_amdgcn_wmma_f32_16x16x32_bf16(
                false, afr[kk], false, bfr[kk], (short)0, acc, false, false);

        // epilogue: D-frag element r -> (m = r + 8*half, n = nlo)
        const int c  = nt * 16 + nlo;
        const float qc  = qrow[c];
        const float bec = be[c];
        const int   mb  = m0 + half * 8;
        #pragma unroll
        for (int r = 0; r < 8; ++r) {
            int   j  = mb + r;
            float e  = acc[r] + bec;
            float kv = krow[(size_t)j * DIM + c];
            float at = qc * kv * inv_sqrt_dk * (e + 1.0f) * e;
            smE[j * DIM + c] = (__bf16)at;   // only own wave's rows
        }
    }

    // ---- GEMM2: edge_out = attn x Woe^T + boe
    //      (reads only this wave's smE rows -> DS in-order, no barrier needed)
    #pragma unroll
    for (int kk = 0; kk < 8; ++kk)
        afr[kk] = load_frag(smE + (m0 + nlo) * DIM + kk * 32 + half * 8);

    float* eout = edge_out + (size_t)bi * NN * DIM;
    #pragma unroll 2
    for (int nt = 0; nt < 16; ++nt) {
        v16bf bfr[8];
        #pragma unroll
        for (int kk = 0; kk < 8; ++kk)
            bfr[kk] = load_frag(smW2 + (nt * 16 + nlo) * DIM + kk * 32 + half * 8);
        v8f acc = {};
        #pragma unroll
        for (int kk = 0; kk < 8; ++kk)
            acc = __builtin_amdgcn_wmma_f32_16x16x32_bf16(
                false, afr[kk], false, bfr[kk], (short)0, acc, false, false);

        const int   c  = nt * 16 + nlo;
        const float bo = boe[c];
        const int   mb = m0 + half * 8;
        #pragma unroll
        for (int r = 0; r < 8; ++r)
            eout[(size_t)(mb + r) * DIM + c] = acc[r] + bo;
    }
    __syncthreads();   // all attn rows visible for the cross-wave reduction

    // ---- per-channel softmax over j + weighted v sum
    {
        const int c = tid;
        float mx = -3.402823466e38f;
        for (int j = 0; j < NN; ++j)
            mx = fmaxf(mx, (float)smE[j * DIM + c]);
        float s = 0.0f, agg = 0.0f;
        const float* vrow = vws + (size_t)b * NN * DIM + c;
        for (int j = 0; j < NN; ++j) {
            float a = __expf((float)smE[j * DIM + c] - mx);
            s   += a;
            agg += a * vrow[(size_t)j * DIM];
        }
        aggws[(size_t)bi * DIM + c] = agg / s;
    }
}

// ---------------------------------------------------------------------------
// Kernel 3: node_out = node_agg @ Won.T + bon (coalesced via WonT)
// ---------------------------------------------------------------------------
__global__ void __launch_bounds__(DIM) nodeout_kernel(
    const float* __restrict__ aggws, const float* __restrict__ WonT,
    const float* __restrict__ bon, float* __restrict__ out_node)
{
    __shared__ float sa[DIM];
    int r = blockIdx.x, c = threadIdx.x;
    sa[c] = aggws[(size_t)r * DIM + c];
    __syncthreads();
    float acc = bon[c];
    #pragma unroll 4
    for (int cc = 0; cc < DIM; ++cc)
        acc += sa[cc] * WonT[cc * DIM + c];
    out_node[(size_t)r * DIM + c] = acc;
}

// ---------------------------------------------------------------------------
extern "C" void kernel_launch(void* const* d_in, const int* in_sizes, int n_in,
                              void* d_out, int out_size, void* d_ws, size_t ws_size,
                              hipStream_t stream) {
    const float* node = (const float*)d_in[0];
    const float* edge = (const float*)d_in[1];
    const float* Wq  = (const float*)d_in[2];  const float* bq  = (const float*)d_in[3];
    const float* Wk  = (const float*)d_in[4];  const float* bk  = (const float*)d_in[5];
    const float* Wv  = (const float*)d_in[6];  const float* bv  = (const float*)d_in[7];
    const float* We  = (const float*)d_in[8];  const float* be  = (const float*)d_in[9];
    const float* Woe = (const float*)d_in[10]; const float* boe = (const float*)d_in[11];
    const float* Won = (const float*)d_in[12]; const float* bon = (const float*)d_in[13];

    float* out_node = (float*)d_out;                       // [16,128,256]
    float* out_edge = out_node + (size_t)BB * NN * DIM;    // [16,128,128,256]

    // workspace layout
    char* ws = (char*)d_ws;
    float* qws   = (float*)(ws);                               // 2 MB
    float* kws   = (float*)(ws + (size_t)(2  << 20));          // 2 MB
    float* vws   = (float*)(ws + (size_t)(4  << 20));          // 2 MB
    float* aggws = (float*)(ws + (size_t)(6  << 20));          // 2 MB
    float* WqT   = (float*)(ws + (size_t)(8  << 20));          // 256 KB each
    float* WkT   = WqT + DIM * DIM;
    float* WvT   = WkT + DIM * DIM;
    float* WonT  = WvT + DIM * DIM;
    __bf16* Webf  = (__bf16*)(WonT + DIM * DIM);               // 128 KB
    __bf16* Woebf = Webf + DIM * DIM;                          // 128 KB

    prep_kernel<<<DIM, DIM, 0, stream>>>(Wq, Wk, Wv, Won, We, Woe,
                                         WqT, WkT, WvT, WonT, Webf, Woebf);
    qkv_kernel<<<ROWS, DIM, 0, stream>>>(node, WqT, bq, WkT, bk, WvT, bv,
                                         qws, kws, vws);
    const size_t lds_bytes = 131072 + 131072 + 65536;   // 320 KB, WGP max
    main_kernel<<<ROWS, DIM, lds_bytes, stream>>>(edge, qws, kws, vws, be, boe,
                                                  Webf, Woebf, aggws, out_edge);
    nodeout_kernel<<<ROWS, DIM, 0, stream>>>(aggws, WonT, bon, out_node);
}
